// MLPPerTaskNoIndex_10411000726250
// MI455X (gfx1250) — matrix operand order repaired and checked by reading, not compiled, not run on hardware
//
#include <hip/hip_runtime.h>

constexpr int NTASK = 16;
constexpr int DIN   = 512;
constexpr int DHID  = 1024;
constexpr int DOUT  = 256;
constexpr int NBATCH = 8192;
constexpr int MT    = 32;                      // rows per block tile (2 WMMA row-tiles)
constexpr int MAXT  = NBATCH / MT + NTASK;     // 272 worst-case tiles

typedef __attribute__((ext_vector_type(16))) __bf16       bf16x16;
typedef __attribute__((ext_vector_type(8)))  float        f32x8;
typedef __attribute__((ext_vector_type(4)))  unsigned int u32x4;
typedef __attribute__((ext_vector_type(2)))  unsigned int u32x2;
typedef __attribute__((ext_vector_type(4)))  float        f32x4;

union ABPack { bf16x16 v; u32x4 q[2]; };

__device__ __forceinline__ unsigned short f2bf(float f) {
  unsigned int u = __float_as_uint(f);
  u += 0x7fffu + ((u >> 16) & 1u);            // round-to-nearest-even
  return (unsigned short)(u >> 16);
}

__device__ __forceinline__ float fast_tanh(float x) {
#if __has_builtin(__builtin_amdgcn_tanhf)
  return __builtin_amdgcn_tanhf(x);
#elif __has_builtin(__builtin_amdgcn_tanh_f32)
  return __builtin_amdgcn_tanh_f32(x);
#else
  return tanhf(x);
#endif
}

__device__ __forceinline__ f32x8 wmma_bf16(bf16x16 a, bf16x16 b, f32x8 c) {
  return __builtin_amdgcn_wmma_f32_16x16x32_bf16(false, a, false, b, (short)0, c,
                                                 false, false);
}

// A fragment (16x32 bf16): lane holds row M=lane&15; elems 0..7 at p, 8..15 at p+16
__device__ __forceinline__ bf16x16 loadA(const unsigned short* p) {
  ABPack a;
  a.q[0] = *(const u32x4*)(p);
  a.q[1] = *(const u32x4*)(p + 16);
  return a.v;
}
// B fragment (32x16 bf16, pre-packed): 16 consecutive ushorts per lane
__device__ __forceinline__ bf16x16 loadB(const unsigned short* p) {
  ABPack b;
  b.q[0] = *(const u32x4*)(p);
  b.q[1] = *(const u32x4*)(p + 8);
  return b.v;
}

__device__ __forceinline__ f32x8 zero8() {
  f32x8 z = {0.f, 0.f, 0.f, 0.f, 0.f, 0.f, 0.f, 0.f};
  return z;
}

// 2x2 tile GEMM inner loop: rows {a0,a1} x cols {w0t,w1t}
__device__ __forceinline__ void gemm2x2(const unsigned short* a0,
                                        const unsigned short* a1,
                                        const unsigned short* w0t,
                                        const unsigned short* w1t, int ksteps,
                                        f32x8& c00, f32x8& c01, f32x8& c10,
                                        f32x8& c11) {
#pragma unroll 2
  for (int kb = 0; kb < ksteps; ++kb) {
    bf16x16 A0 = loadA(a0 + kb * 32);
    bf16x16 A1 = loadA(a1 + kb * 32);
    bf16x16 B0 = loadB(w0t + kb * 512);
    bf16x16 B1 = loadB(w1t + kb * 512);
    c00 = wmma_bf16(A0, B0, c00);
    c01 = wmma_bf16(A0, B1, c01);
    c10 = wmma_bf16(A1, B0, c10);
    c11 = wmma_bf16(A1, B1, c11);
  }
}

// ---------------- sort-by-task machinery ----------------
__global__ void init_counts(int* counts) {
  if (threadIdx.x < NTASK) counts[threadIdx.x] = 0;
}

__global__ void hist_kernel(const int* __restrict__ tids, int* __restrict__ counts) {
  int i = blockIdx.x * 256 + threadIdx.x;
  if (i < NBATCH) atomicAdd(&counts[tids[i]], 1);
}

__global__ void scan_kernel(const int* __restrict__ counts, int* __restrict__ offsets,
                            int* __restrict__ cursors, int* __restrict__ tileTask,
                            int* __restrict__ tileRow, int* __restrict__ numTiles) {
  if (threadIdx.x != 0 || blockIdx.x != 0) return;
  int off = 0, nt = 0;
  for (int t = 0; t < NTASK; ++t) {
    offsets[t] = off;
    cursors[t] = off;
    int c = counts[t];
    for (int s = 0; s < c; s += MT) { tileTask[nt] = t; tileRow[nt] = off + s; ++nt; }
    off += c;
  }
  offsets[NTASK] = off;
  *numTiles = nt;
}

__global__ void scatter_kernel(const int* __restrict__ tids, int* __restrict__ cursors,
                               int* __restrict__ perm) {
  int i = blockIdx.x * 256 + threadIdx.x;
  if (i < NBATCH) {
    int p = atomicAdd(&cursors[tids[i]], 1);
    perm[p] = i;
  }
}

// gather x rows into task-sorted order, f32 -> bf16
__global__ void gather_x(const float* __restrict__ x, const int* __restrict__ perm,
                         unsigned short* __restrict__ xg) {
  int tid = blockIdx.x * 256 + threadIdx.x;        // NBATCH * (DIN/4) threads
  int sr = tid >> 7;
  int c  = (tid & 127) << 2;
  int src = perm[sr];
  f32x4 v = *(const f32x4*)(x + (size_t)src * DIN + c);
  u32x2 o;
  o.x = (unsigned int)f2bf(v.x) | ((unsigned int)f2bf(v.y) << 16);
  o.y = (unsigned int)f2bf(v.z) | ((unsigned int)f2bf(v.w) << 16);
  *(u32x2*)(xg + (size_t)sr * DIN + c) = o;
}

// ---------------- weight packing: W[t][K][N] f32 -> WMMA B-fragment bf16 layout ----
// Packed layout per task: tiles (nb, kb) of 32(K)x16(N); within a tile, element
// index = lane*16 + e with lane = 16*h + (n%16), k%32 = 8h+e (e<8), 16+8h+(e-8) (e>=8).
// Tile order: nb-major, kb-minor -> k-streaming is contiguous per n-tile.
__global__ __launch_bounds__(256) void pack_weights(const float* __restrict__ src,
                                                    unsigned short* __restrict__ dst,
                                                    int K, int N) {
  __shared__ float tile[64][68];
  int t = blockIdx.y;
  int kTiles64 = K >> 6;
  int kb0 = blockIdx.x % kTiles64;
  int nb0 = blockIdx.x / kTiles64;
  int tid = threadIdx.x;

  const float* s = src + ((size_t)t * K + (size_t)kb0 * 64) * N + (size_t)nb0 * 64;
  for (int i = 0; i < 4; ++i) {
    int idx = tid + 256 * i;          // 1024 float4s
    int r = idx >> 4;
    int c = (idx & 15) << 2;
    f32x4 v = *(const f32x4*)(s + (size_t)r * N + c);
    tile[r][c] = v.x; tile[r][c + 1] = v.y; tile[r][c + 2] = v.z; tile[r][c + 3] = v.w;
  }
  __syncthreads();

  unsigned int* d32 = (unsigned int*)dst;
  size_t tbase = (size_t)t * K * N;
  int ksteps = K >> 5;
  for (int i = 0; i < 8; ++i) {                 // 8 packed 32x16 tiles in this 64x64
    int kb = i >> 2, nb = i & 3;
    int e0   = (2 * tid) & 15;
    int lane = (2 * tid) >> 4;                  // 0..31
    int kl   = ((e0 >> 3) << 4) + ((lane >> 4) << 3) + (e0 & 7);  // k within 32-row tile
    int nl   = lane & 15;
    float v0 = tile[kb * 32 + kl][nb * 16 + nl];
    float v1 = tile[kb * 32 + kl + 1][nb * 16 + nl];
    unsigned int pk = (unsigned int)f2bf(v0) | ((unsigned int)f2bf(v1) << 16);
    size_t off = tbase + ((size_t)(nb0 * 4 + nb) * ksteps + (kb0 * 2 + kb)) * 512;
    d32[off / 2 + tid] = pk;
  }
}

// ---------------- fused 3-layer MLP, one 32-row task tile per block -------------
__global__ __launch_bounds__(256) void mlp_kernel(
    const unsigned short* __restrict__ xg,
    const unsigned short* __restrict__ w0,
    const unsigned short* __restrict__ w1,
    const unsigned short* __restrict__ w2,
    const float* __restrict__ b0,
    const float* __restrict__ b1,
    const float* __restrict__ b2,
    const int* __restrict__ perm,
    const int* __restrict__ offsets,
    const int* __restrict__ tileTask,
    const int* __restrict__ tileRow,
    const int* __restrict__ numTiles,
    float* __restrict__ out) {
  __shared__ __align__(16) unsigned short h0[MT * DHID];   // 64 KB
  __shared__ __align__(16) unsigned short h1[MT * DHID];   // 64 KB

  int bid = blockIdx.x;
  if (bid >= *numTiles) return;
  int task = tileTask[bid];
  int rowStart = tileRow[bid];
  int rowEnd = offsets[task + 1];

  int tid  = threadIdx.x;
  int lane = tid & 31;
  int wave = tid >> 5;
  int kh   = lane >> 4;        // K-half for A fragments
  int nl   = lane & 15;        // N column (B/C) == A row within row-tile

  int aRow0 = rowStart + nl;
  int aRow1 = rowStart + 16 + nl;
  if (aRow0 >= rowEnd) aRow0 = rowEnd - 1;     // clamp padded rows
  if (aRow1 >= rowEnd) aRow1 = rowEnd - 1;
  const unsigned short* xr0 = xg + (size_t)aRow0 * DIN + 8 * kh;
  const unsigned short* xr1 = xg + (size_t)aRow1 * DIN + 8 * kh;

  // ---- layer 0: [32 x 512] x [512 x 1024] + b0 -> tanh -> h0 (LDS)
  {
    const unsigned short* wt = w0 + (size_t)task * DIN * DHID + lane * 16;
    const float* bias = b0 + task * DHID;
    const int ksteps = DIN / 32;                         // 16
    for (int it = 0; it < 4; ++it) {                     // 4 nb-pairs per wave
      int nb0 = wave + 8 * it;                           // 0..31
      int nb1 = nb0 + 32;                                // 32..63
      f32x8 c00 = zero8(), c01 = zero8(), c10 = zero8(), c11 = zero8();
      gemm2x2(xr0, xr1, wt + (size_t)nb0 * ksteps * 512,
              wt + (size_t)nb1 * ksteps * 512, ksteps, c00, c01, c10, c11);
      float bv0 = bias[nb0 * 16 + nl];
      float bv1 = bias[nb1 * 16 + nl];
#pragma unroll
      for (int r = 0; r < 8; ++r) {
        int m = 8 * kh + r;
        h0[m * DHID + nb0 * 16 + nl]        = f2bf(fast_tanh(c00[r] + bv0));
        h0[m * DHID + nb1 * 16 + nl]        = f2bf(fast_tanh(c01[r] + bv1));
        h0[(m + 16) * DHID + nb0 * 16 + nl] = f2bf(fast_tanh(c10[r] + bv0));
        h0[(m + 16) * DHID + nb1 * 16 + nl] = f2bf(fast_tanh(c11[r] + bv1));
      }
    }
  }
  __syncthreads();

  // ---- layer 1: [32 x 1024] x [1024 x 1024] + b1 -> tanh -> h1 (LDS)
  {
    const unsigned short* wt = w1 + (size_t)task * DHID * DHID + lane * 16;
    const float* bias = b1 + task * DHID;
    const int ksteps = DHID / 32;                        // 32
    const unsigned short* a0 = h0 + nl * DHID + 8 * kh;
    const unsigned short* a1 = h0 + (16 + nl) * DHID + 8 * kh;
    for (int it = 0; it < 4; ++it) {
      int nb0 = wave + 8 * it;
      int nb1 = nb0 + 32;
      f32x8 c00 = zero8(), c01 = zero8(), c10 = zero8(), c11 = zero8();
      gemm2x2(a0, a1, wt + (size_t)nb0 * ksteps * 512,
              wt + (size_t)nb1 * ksteps * 512, ksteps, c00, c01, c10, c11);
      float bv0 = bias[nb0 * 16 + nl];
      float bv1 = bias[nb1 * 16 + nl];
#pragma unroll
      for (int r = 0; r < 8; ++r) {
        int m = 8 * kh + r;
        h1[m * DHID + nb0 * 16 + nl]        = f2bf(fast_tanh(c00[r] + bv0));
        h1[m * DHID + nb1 * 16 + nl]        = f2bf(fast_tanh(c01[r] + bv1));
        h1[(m + 16) * DHID + nb0 * 16 + nl] = f2bf(fast_tanh(c10[r] + bv0));
        h1[(m + 16) * DHID + nb1 * 16 + nl] = f2bf(fast_tanh(c11[r] + bv1));
      }
    }
  }
  __syncthreads();

  // ---- layer 2: [32 x 1024] x [1024 x 256] + b2 -> out (f32, original row order)
  {
    const unsigned short* wt = w2 + (size_t)task * DHID * DOUT + lane * 16;
    const float* bias = b2 + task * DOUT;
    const int ksteps = DHID / 32;                        // 32
    const unsigned short* a0 = h1 + nl * DHID + 8 * kh;
    const unsigned short* a1 = h1 + (16 + nl) * DHID + 8 * kh;
    {
      int nb0 = wave;                                    // 0..7
      int nb1 = wave + 8;                                // 8..15
      f32x8 c00 = zero8(), c01 = zero8(), c10 = zero8(), c11 = zero8();
      gemm2x2(a0, a1, wt + (size_t)nb0 * ksteps * 512,
              wt + (size_t)nb1 * ksteps * 512, ksteps, c00, c01, c10, c11);
      float bv0 = bias[nb0 * 16 + nl];
      float bv1 = bias[nb1 * 16 + nl];
#pragma unroll
      for (int r = 0; r < 8; ++r) {
        int m = 8 * kh + r;
        int sr0 = rowStart + m;
        int sr1 = rowStart + 16 + m;
        if (sr0 < rowEnd) {
          int orow = perm[sr0];
          out[(size_t)orow * DOUT + nb0 * 16 + nl] = c00[r] + bv0;
          out[(size_t)orow * DOUT + nb1 * 16 + nl] = c01[r] + bv1;
        }
        if (sr1 < rowEnd) {
          int orow = perm[sr1];
          out[(size_t)orow * DOUT + nb0 * 16 + nl] = c10[r] + bv0;
          out[(size_t)orow * DOUT + nb1 * 16 + nl] = c11[r] + bv1;
        }
      }
    }
  }
}

extern "C" void kernel_launch(void* const* d_in, const int* in_sizes, int n_in,
                              void* d_out, int out_size, void* d_ws, size_t ws_size,
                              hipStream_t stream) {
  const float* x   = (const float*)d_in[0];
  const int*   tid = (const int*)d_in[1];
  const float* k0  = (const float*)d_in[2];
  const float* b0  = (const float*)d_in[3];
  const float* k1  = (const float*)d_in[4];
  const float* b1  = (const float*)d_in[5];
  const float* k2  = (const float*)d_in[6];
  const float* b2  = (const float*)d_in[7];
  float* out = (float*)d_out;

  char* ws = (char*)d_ws;
  size_t off = 0;
  auto carve = [&](size_t bytes) -> void* {
    void* p = ws + off;
    off += (bytes + 255) & ~(size_t)255;
    return p;
  };
  unsigned short* w0p = (unsigned short*)carve((size_t)NTASK * DIN * DHID * 2);
  unsigned short* w1p = (unsigned short*)carve((size_t)NTASK * DHID * DHID * 2);
  unsigned short* w2p = (unsigned short*)carve((size_t)NTASK * DHID * DOUT * 2);
  unsigned short* xgp = (unsigned short*)carve((size_t)NBATCH * DIN * 2);
  int* perm     = (int*)carve((size_t)NBATCH * 4);
  int* counts   = (int*)carve(NTASK * 4);
  int* cursors  = (int*)carve(NTASK * 4);
  int* offsets  = (int*)carve((NTASK + 1) * 4);
  int* tileTask = (int*)carve(MAXT * 4);
  int* tileRow  = (int*)carve(MAXT * 4);
  int* numTiles = (int*)carve(4);

  init_counts<<<1, 32, 0, stream>>>(counts);
  hist_kernel<<<NBATCH / 256, 256, 0, stream>>>(tid, counts);
  scan_kernel<<<1, 1, 0, stream>>>(counts, offsets, cursors, tileTask, tileRow, numTiles);
  scatter_kernel<<<NBATCH / 256, 256, 0, stream>>>(tid, cursors, perm);
  gather_x<<<(NBATCH * (DIN / 4)) / 256, 256, 0, stream>>>(x, perm, xgp);

  pack_weights<<<dim3((DIN / 64) * (DHID / 64), NTASK), 256, 0, stream>>>(k0, w0p, DIN, DHID);
  pack_weights<<<dim3((DHID / 64) * (DHID / 64), NTASK), 256, 0, stream>>>(k1, w1p, DHID, DHID);
  pack_weights<<<dim3((DHID / 64) * (DOUT / 64), NTASK), 256, 0, stream>>>(k2, w2p, DHID, DOUT);

  mlp_kernel<<<MAXT, 256, 0, stream>>>(xgp, w0p, w1p, w2p, b0, b1, b2, perm,
                                       offsets, tileTask, tileRow, numTiles, out);
}
